// Net_55533927137869
// MI455X (gfx1250) — matrix operand order repaired and checked by reading, not compiled, not run on hardware
//
#include <hip/hip_runtime.h>

// ---------------------------------------------------------------------------
// Types for CDNA5 WMMA (wave32): v_wmma_f32_16x16x32_bf16
// ---------------------------------------------------------------------------
typedef __bf16 bf16_t;
typedef bf16_t v16bf __attribute__((ext_vector_type(16)));
typedef float  v8f   __attribute__((ext_vector_type(8)));

static __device__ __forceinline__ v8f wmma_bf16(v16bf a, v16bf b, v8f c) {
  // 8 args: (neg_a, A, neg_b, B, c_mod, C, reuse_a, reuse_b)
  return __builtin_amdgcn_wmma_f32_16x16x32_bf16(false, a, false, b, (short)0, c,
                                                 false, false);
}

// Fast transcendentals: CDNA5 has v_tanh_f32 / v_exp_f32 / v_rcp_f32 in the
// TRANS pipe; avoid the branchy libm expansions between WMMA clusters.
static __device__ __forceinline__ float fast_tanh(float x) {
#if __has_builtin(__builtin_amdgcn_tanhf)
  return __builtin_amdgcn_tanhf(x);
#else
  return tanhf(x);
#endif
}
static __device__ __forceinline__ float fast_sigmoid(float x) {
  float e = __expf(-x);
#if __has_builtin(__builtin_amdgcn_rcpf)
  return __builtin_amdgcn_rcpf(1.0f + e);
#else
  return 1.0f / (1.0f + e);
#endif
}

union pk8 { bf16_t h[8]; uint4 u; };

// ---------------------------------------------------------------------------
// Pack fp32 weights into WMMA A-fragment layout (16x32 bf16 tile, per-lane
// 16 contiguous bf16).  A layout used here (contiguous k; B uses the same):
//   lanes 0-15 : M=lane,    K=j      (0..15)
//   lanes16-31 : M=lane-16, K=16+j   (16..31)
// Fragment classes (blockIdx ranges):
//   [0,162)    Wt    : 27 layers x 3 taps x 2 mtiles, src [L][32][32][3]
//   [162,324)  Ws    : same shape
//   [324,378)  Wd    : 27 x 2,              src [L][32][32][1]
//   [378,1242) Wskip : idx = mt*27+i (mt<32), src [L][512][32][1]
//   [1242,1754)Wp1   : idx = mt*16+kt (mt<32,kt<16), src [512][512][1]
//   [1754,2010)Wp2   : idx = mt*16+kt (mt<16,kt<16), src [256][512][1]
// ---------------------------------------------------------------------------
__global__ void k_pack_all(const float* __restrict__ Wt, const float* __restrict__ Ws,
                           const float* __restrict__ Wd, const float* __restrict__ Wsk,
                           const float* __restrict__ W1, const float* __restrict__ W2,
                           bf16_t* WtP, bf16_t* WsP, bf16_t* WdP,
                           bf16_t* WskP, bf16_t* W1P, bf16_t* W2P)
{
  int fid  = blockIdx.x;
  int lane = threadIdx.x;
  int mloc = lane & 15;
  int kb   = (lane < 16) ? 0 : 16;

  const float* src; bf16_t* dst;
  long strideM, strideK, off; int m0;
  if (fid < 162) {
    int f = fid, i = f / 6, r = f % 6, tap = r >> 1, mt = r & 1;
    src = Wt; dst = WtP + (size_t)f * 512; m0 = mt * 16;
    strideM = 32 * 3; strideK = 3; off = (long)i * 32 * 32 * 3 + tap;
  } else if (fid < 324) {
    int f = fid - 162, i = f / 6, r = f % 6, tap = r >> 1, mt = r & 1;
    src = Ws; dst = WsP + (size_t)f * 512; m0 = mt * 16;
    strideM = 32 * 3; strideK = 3; off = (long)i * 32 * 32 * 3 + tap;
  } else if (fid < 378) {
    int f = fid - 324, i = f / 2, mt = f & 1;
    src = Wd; dst = WdP + (size_t)f * 512; m0 = mt * 16;
    strideM = 32; strideK = 1; off = (long)i * 32 * 32;
  } else if (fid < 1242) {
    int f = fid - 378, mt = f / 27, i = f % 27;
    src = Wsk; dst = WskP + (size_t)f * 512; m0 = mt * 16;
    strideM = 32; strideK = 1; off = (long)i * 512 * 32;
  } else if (fid < 1754) {
    int f = fid - 1242, mt = f / 16, kt = f % 16;
    src = W1; dst = W1P + (size_t)f * 512; m0 = mt * 16;
    strideM = 512; strideK = 1; off = (long)kt * 32;
  } else {
    int f = fid - 1754, mt = f / 16, kt = f % 16;
    src = W2; dst = W2P + (size_t)f * 512; m0 = mt * 16;
    strideM = 512; strideK = 1; off = (long)kt * 32;
  }
  int m = m0 + mloc;
#pragma unroll
  for (int j = 0; j < 16; ++j) {
    int kk = kb + j;
    float v = src[off + (long)m * strideM + (long)kk * strideK];
    dst[(size_t)lane * 16 + j] = (bf16_t)v;
  }
}

// sum of the 27 skip biases (each layer adds its bias to skip)
__global__ void k_bias_sum(const float* __restrict__ bskip, float* __restrict__ bsum) {
  int c = blockIdx.x * blockDim.x + threadIdx.x;
  if (c < 512) { float s = 0.f; for (int i = 0; i < 27; ++i) s += bskip[i * 512 + c]; bsum[c] = s; }
}

// front conv: 1 -> 32 channels, kernel 3, padding 1; output X0[t][c] bf16
__global__ void k_front(const float* __restrict__ xin, const float* __restrict__ Wc,
                        const float* __restrict__ bc, bf16_t* __restrict__ X0, int T) {
  int idx = blockIdx.x * blockDim.x + threadIdx.x;
  int t = idx >> 5, c = idx & 31;
  if (t >= T) return;
  float xm = (t > 0) ? xin[t - 1] : 0.f;
  float x0 = xin[t];
  float xp = (t + 1 < T) ? xin[t + 1] : 0.f;
  float v = bc[c] + Wc[c * 3 + 0] * xm + Wc[c * 3 + 1] * x0 + Wc[c * 3 + 2] * xp;
  X0[(size_t)t * 32 + c] = (bf16_t)v;
}

// ---------------------------------------------------------------------------
// One residual layer.  block = 256 threads = 8 waves; each wave owns 16 time
// columns.  The 14 per-layer A-fragments (Wt:6, Ws:6, Wdense:2; 14 KB) are
// staged once per block into LDS and shared by all 8 waves, so the inner
// GEMMs read A via ds_load instead of re-fetching from L2 per wave.
// Gates: 12 WMMA; dense: 2 WMMA.  Gated activation g is written to the
// deferred-skip buffer Gout (bf16) and round-tripped through LDS to become a
// B-fragment for the dense 1x1.
// ---------------------------------------------------------------------------
__global__ void __launch_bounds__(256) k_layer(
    const bf16_t* __restrict__ Xin, bf16_t* __restrict__ Xout,
    const bf16_t* __restrict__ AtP, const bf16_t* __restrict__ AsP,
    const bf16_t* __restrict__ AdP,
    const float* __restrict__ bt, const float* __restrict__ bs,
    const float* __restrict__ bd,
    bf16_t* __restrict__ Gout, int d, int T)
{
  __shared__ __attribute__((aligned(32))) bf16_t aLds[14 * 512];
  __shared__ __attribute__((aligned(32))) bf16_t gsh[8][16][32];
  int tid  = threadIdx.x;

  // cooperative stage of 14 fragments (14*512 bf16 = 896 uint4 chunks)
  for (int e = tid; e < 896; e += 256) {
    int frag = e >> 6;          // 64 chunks per fragment
    int c    = e & 63;
    const bf16_t* src = (frag < 6)  ? (AtP + (size_t)frag * 512)
                      : (frag < 12) ? (AsP + (size_t)(frag - 6) * 512)
                                    : (AdP + (size_t)(frag - 12) * 512);
    *(uint4*)&aLds[(size_t)e * 8] = *(const uint4*)(src + (size_t)c * 8);
  }
  __syncthreads();

  int w    = tid >> 5, lane = tid & 31;
  int n    = lane & 15;
  int hi   = lane >> 4;
  int kb16 = hi * 16;
  int kb8  = hi * 8;
  int t  = blockIdx.x * 128 + w * 16 + n;
  int tm = t - d, tp = t + d;

  // B fragments for the three taps (32 ch x 16 cols), zero pad at edges
  v16bf b0 = *(const v16bf*)(Xin + (size_t)t * 32 + kb16);
  v16bf bm = {}; if (tm >= 0) bm = *(const v16bf*)(Xin + (size_t)tm * 32 + kb16);
  v16bf bp = {}; if (tp < T)  bp = *(const v16bf*)(Xin + (size_t)tp * 32 + kb16);
  v16bf btap[3] = {bm, b0, bp};

  v8f zf = {};
  v8f accT0 = zf, accT1 = zf, accS0 = zf, accS1 = zf;
#pragma unroll
  for (int p = 0; p < 3; ++p) {
    accT0 = wmma_bf16(*(const v16bf*)&aLds[(size_t)(p * 2 + 0) * 512 + (size_t)lane * 16], btap[p], accT0);
    accT1 = wmma_bf16(*(const v16bf*)&aLds[(size_t)(p * 2 + 1) * 512 + (size_t)lane * 16], btap[p], accT1);
    accS0 = wmma_bf16(*(const v16bf*)&aLds[(size_t)(6 + p * 2 + 0) * 512 + (size_t)lane * 16], btap[p], accS0);
    accS1 = wmma_bf16(*(const v16bf*)&aLds[(size_t)(6 + p * 2 + 1) * 512 + (size_t)lane * 16], btap[p], accS1);
  }

  // gate, write g to LDS (for dense B-frag) and to global G (deferred skip)
#pragma unroll
  for (int mt = 0; mt < 2; ++mt) {
    v8f aT = mt ? accT1 : accT0;
    v8f aS = mt ? accS1 : accS0;
    int c0 = mt * 16 + kb8;
    pk8 pk;
#pragma unroll
    for (int r = 0; r < 8; ++r) {
      int c = c0 + r;
      float gt = fast_tanh(aT[r] + bt[c]);
      float gs = fast_sigmoid(aS[r] + bs[c]);
      pk.h[r] = (bf16_t)(gt * gs);
    }
    *(uint4*)&gsh[w][n][c0] = pk.u;
    *(uint4*)(Gout + (size_t)t * 32 + c0) = pk.u;
  }

  // dense 1x1 + residual
  v16bf bg = *(const v16bf*)&gsh[w][n][kb16];
#pragma unroll
  for (int mt = 0; mt < 2; ++mt) {
    v8f ad = zf;
    ad = wmma_bf16(*(const v16bf*)&aLds[(size_t)(12 + mt) * 512 + (size_t)lane * 16], bg, ad);
    int c0 = mt * 16 + kb8;
    pk8 xi; xi.u = *(const uint4*)(Xin + (size_t)t * 32 + c0);
    pk8 po;
#pragma unroll
    for (int r = 0; r < 8; ++r)
      po.h[r] = (bf16_t)(ad[r] + bd[c0 + r] + (float)xi.h[r]);
    *(uint4*)(Xout + (size_t)t * 32 + c0) = po.u;
  }
}

// ---------------------------------------------------------------------------
// Post net, fused: skip = Wsk_all[512x864] @ G[864xT]  (deferred-skip GEMM),
// h = relu(Wp1 @ skip + bp1), o = Wp2 @ h + bp2, log_softmax over 256 ch.
// block = 256 threads = 8 waves handles 16 time columns.
// ---------------------------------------------------------------------------
__global__ void __launch_bounds__(256) k_post(
    const bf16_t* __restrict__ G, const bf16_t* __restrict__ AskP,
    const bf16_t* __restrict__ A1P, const bf16_t* __restrict__ A2P,
    const float* __restrict__ bsum, const float* __restrict__ bp1,
    const float* __restrict__ bp2, float* __restrict__ out, int T)
{
  __shared__ __attribute__((aligned(32))) bf16_t sSkip[16][512];
  __shared__ __attribute__((aligned(32))) bf16_t sH[16][512];
  __shared__ float sOut[16][256];
  __shared__ float sRed[16][16];
  __shared__ float sVal[16];

  int tid  = threadIdx.x;
  int w    = tid >> 5, lane = tid & 31;
  int n    = lane & 15, hi = lane >> 4;
  int kb16 = hi * 16, kb8 = hi * 8;
  int t0 = blockIdx.x * 16;
  int t  = t0 + n;
  v8f zf = {};

  // Phase 1: skip GEMM  (M=512 -> 32 mtiles, 4 per wave; K = 27 x 32)
  v8f acc[4] = {zf, zf, zf, zf};
  for (int i = 0; i < 27; ++i) {
    v16bf bg = *(const v16bf*)(G + ((size_t)i * T + t) * 32 + kb16);
#pragma unroll
    for (int j = 0; j < 4; ++j) {
      int mt = w * 4 + j;
      acc[j] = wmma_bf16(*(const v16bf*)(AskP + ((size_t)mt * 27 + i) * 512 + (size_t)lane * 16), bg, acc[j]);
    }
  }
#pragma unroll
  for (int j = 0; j < 4; ++j) {
    int c0 = (w * 4 + j) * 16 + kb8;
#pragma unroll
    for (int r = 0; r < 8; ++r)
      sSkip[n][c0 + r] = (bf16_t)(acc[j][r] + bsum[c0 + r]);
  }
  __syncthreads();

  // Phase 2: h = relu(Wp1 @ skip + bp1)   (M=512, K=512)
#pragma unroll
  for (int j = 0; j < 4; ++j) acc[j] = zf;
  for (int kt = 0; kt < 16; ++kt) {
    v16bf bsk = *(const v16bf*)&sSkip[n][kt * 32 + kb16];
#pragma unroll
    for (int j = 0; j < 4; ++j) {
      int mt = w * 4 + j;
      acc[j] = wmma_bf16(*(const v16bf*)(A1P + ((size_t)mt * 16 + kt) * 512 + (size_t)lane * 16), bsk, acc[j]);
    }
  }
#pragma unroll
  for (int j = 0; j < 4; ++j) {
    int c0 = (w * 4 + j) * 16 + kb8;
#pragma unroll
    for (int r = 0; r < 8; ++r)
      sH[n][c0 + r] = (bf16_t)fmaxf(acc[j][r] + bp1[c0 + r], 0.f);
  }
  __syncthreads();

  // Phase 3: o = Wp2 @ h + bp2   (M=256 -> 16 mtiles, 2 per wave)
  v8f acc2[2] = {zf, zf};
  for (int kt = 0; kt < 16; ++kt) {
    v16bf bh = *(const v16bf*)&sH[n][kt * 32 + kb16];
#pragma unroll
    for (int j = 0; j < 2; ++j) {
      int mt = w * 2 + j;
      acc2[j] = wmma_bf16(*(const v16bf*)(A2P + ((size_t)mt * 16 + kt) * 512 + (size_t)lane * 16), bh, acc2[j]);
    }
  }
#pragma unroll
  for (int j = 0; j < 2; ++j) {
    int c0 = (w * 2 + j) * 16 + kb8;
#pragma unroll
    for (int r = 0; r < 8; ++r)
      sOut[n][c0 + r] = acc2[j][r] + bp2[c0 + r];
  }
  __syncthreads();

  // Phase 4: log_softmax over the 256 channels of each column
  int col = tid & 15, part = tid >> 4;
  float mx = -3.4e38f;
#pragma unroll
  for (int u = 0; u < 16; ++u) mx = fmaxf(mx, sOut[col][part * 16 + u]);
  sRed[col][part] = mx;
  __syncthreads();
  if (part == 0) {
    float m = -3.4e38f;
#pragma unroll
    for (int p = 0; p < 16; ++p) m = fmaxf(m, sRed[col][p]);
    sVal[col] = m;
  }
  __syncthreads();
  float m = sVal[col];
  float s = 0.f;
#pragma unroll
  for (int u = 0; u < 16; ++u) s += __expf(sOut[col][part * 16 + u] - m);
  sRed[col][part] = s;
  __syncthreads();
  if (part == 0) {
    float ss = 0.f;
#pragma unroll
    for (int p = 0; p < 16; ++p) ss += sRed[col][p];
    sVal[col] = m + __logf(ss);
  }
  __syncthreads();
  float lse = sVal[col];
#pragma unroll
  for (int u = 0; u < 16; ++u)
    out[(size_t)(part * 16 + u) * T + (t0 + col)] = sOut[col][part * 16 + u] - lse;
}

// ---------------------------------------------------------------------------
extern "C" void kernel_launch(void* const* d_in, const int* in_sizes, int n_in,
                              void* d_out, int out_size, void* d_ws, size_t ws_size,
                              hipStream_t stream)
{
  (void)n_in; (void)out_size; (void)ws_size;
  const float* x   = (const float*)d_in[0];
  const float* Wc  = (const float*)d_in[1];
  const float* bc  = (const float*)d_in[2];
  const float* Wt  = (const float*)d_in[3];
  const float* bt  = (const float*)d_in[4];
  const float* Ws  = (const float*)d_in[5];
  const float* bs  = (const float*)d_in[6];
  const float* Wsk = (const float*)d_in[7];
  const float* bsk = (const float*)d_in[8];
  const float* Wd  = (const float*)d_in[9];
  const float* bd  = (const float*)d_in[10];
  const float* W1  = (const float*)d_in[11];
  const float* b1  = (const float*)d_in[12];
  const float* W2  = (const float*)d_in[13];
  const float* b2  = (const float*)d_in[14];
  float* out = (float*)d_out;
  const int T = in_sizes[0];   // 131072
  const int L = 27;

  char* ws = (char*)d_ws;
  size_t cur = 0;
  auto carve = [&](size_t bytes) -> void* {
    void* p = ws + cur;
    cur += (bytes + 255) & ~(size_t)255;
    return p;
  };
  bf16_t* X0   = (bf16_t*)carve((size_t)T * 32 * 2);
  bf16_t* X1   = (bf16_t*)carve((size_t)T * 32 * 2);
  bf16_t* G    = (bf16_t*)carve((size_t)L * T * 32 * 2);
  bf16_t* WtP  = (bf16_t*)carve((size_t)162 * 512 * 2);
  bf16_t* WsP  = (bf16_t*)carve((size_t)162 * 512 * 2);
  bf16_t* WdP  = (bf16_t*)carve((size_t)54 * 512 * 2);
  bf16_t* WskP = (bf16_t*)carve((size_t)864 * 512 * 2);
  bf16_t* W1P  = (bf16_t*)carve((size_t)512 * 512 * 2);
  bf16_t* W2P  = (bf16_t*)carve((size_t)256 * 512 * 2);
  float*  bsum = (float*)carve((size_t)512 * 4);

  k_pack_all<<<2010, 32, 0, stream>>>(Wt, Ws, Wd, Wsk, W1, W2,
                                      WtP, WsP, WdP, WskP, W1P, W2P);
  k_bias_sum<<<2, 256, 0, stream>>>(bsk, bsum);
  k_front<<<(T * 32) / 256, 256, 0, stream>>>(x, Wc, bc, X0, T);

  bf16_t* Xin = X0; bf16_t* Xout = X1;
  for (int i = 0; i < L; ++i) {
    int d = 1 << (i % 9);
    k_layer<<<T / 128, 256, 0, stream>>>(Xin, Xout,
        WtP + (size_t)i * 6 * 512, WsP + (size_t)i * 6 * 512,
        WdP + (size_t)i * 2 * 512,
        bt + i * 32, bs + i * 32, bd + i * 32,
        G + (size_t)i * T * 32, d, T);
    bf16_t* tmp = Xin; Xin = Xout; Xout = tmp;
  }

  k_post<<<T / 16, 256, 0, stream>>>(G, WskP, W1P, W2P, bsum, b1, b2, out, T);
}
